// EdgeNet_edge_45526653337876
// MI455X (gfx1250) — matrix operand order repaired and checked by reading, not compiled, not run on hardware
//
#include <hip/hip_runtime.h>
#include <hip/hip_bf16.h>
#include <stdint.h>

// ---------------------------------------------------------------------------
// CDNA5 (gfx1250) WMMA implementation of the EdgeNet GNN forward pass.
// wave32; V_WMMA_F32_16X16X32_F16 for every per-edge MLP layer.
//  - Weights pre-packed (f16, transposed, zero-padded) into a row-major
//    [KPAD][32] B[k][n] matrix -> B fragments are unguarded strided gathers.
//  - BatchNorm (eval) folded to per-channel scale/shift once per launch ->
//    hot edge kernels do one fmaf per element instead of rsqrt chains.
// ---------------------------------------------------------------------------

typedef __attribute__((ext_vector_type(16))) _Float16 v16h;
typedef __attribute__((ext_vector_type(8)))  _Float16 v8h;
typedef __attribute__((ext_vector_type(8)))  float    v8f;

#define EPSBN 1e-5f

__device__ __forceinline__ v8f wmma16(v16h a, v16h b, v8f c) {
  // D = A(16x32 f16) * B(32x16 f16) + C(16x16 f32)
  return __builtin_amdgcn_wmma_f32_16x16x32_f16(
      /*neg_a=*/false, a, /*neg_b=*/false, b,
      /*c_mod=*/(short)0, c, /*reuse_a=*/false, /*reuse_b=*/false);
}

// A fragment (16-bit A, 16x32 tile) from a 16B-aligned LDS matrix [16][KW].
// ISA 7.12.2: lane&15 = M row; lanes>=16 shift K by +8; per lane the 16
// K-values are two contiguous 8-element runs -> two ds_load_b128 + shuffle.
__device__ __forceinline__ v16h load_A(const _Float16* __restrict__ s,
                                       int lane, int kb, int KW) {
  const int row = lane & 15;
  const int khi = (lane >> 4) << 3;  // 0 or 8
  const _Float16* p = s + row * KW + kb + khi;
  v8h lo = *(const v8h*)(p);         // K = khi .. khi+7
  v8h hi = *(const v8h*)(p + 16);    // K = khi+16 .. khi+23
  return __builtin_shufflevector(lo, hi, 0, 1, 2, 3, 4, 5, 6, 7,
                                 8, 9, 10, 11, 12, 13, 14, 15);
}

// B fragment from a pre-packed matrix P[KPAD][32] (f16, B[k][n]=W[n][k],
// zero padded).  Lane l holds column N = nbase + (l&15); K-values in the same
// order as load_A (khi..khi+7, khi+16..khi+23), strided gather.
__device__ __forceinline__ v16h load_Bp(const _Float16* __restrict__ P,
                                        int kb, int nbase, int lane) {
  const int n = nbase + (lane & 15);
  const int khi = (lane >> 4) << 3;
  v16h b;
#pragma unroll
  for (int s = 0; s < 8; ++s) {
    b[s]     = P[(size_t)(kb + khi + s) * 32 + n];
    b[s + 8] = P[(size_t)(kb + khi + 16 + s) * 32 + n];
  }
  return b;
}

// ---------------------------------------------------------------------------
// One-time packers
// ---------------------------------------------------------------------------
__global__ void pack_w_kernel(const float* __restrict__ W, _Float16* __restrict__ P,
                              int KIN, int DOUT, int KPAD) {
  int i = blockIdx.x * blockDim.x + threadIdx.x;
  if (i < KPAD * 32) {
    int k = i >> 5, n = i & 31;
    float w = (k < KIN && n < DOUT) ? W[n * KIN + k] : 0.0f;
    P[i] = (_Float16)w;
  }
}

// BN(eval): y = (x-mean)*rsqrt(var+eps)*gamma+beta  ->  y = x*sc + sh
__global__ void pack_bn_kernel(const float* __restrict__ g, const float* __restrict__ bt,
                               const float* __restrict__ mn, const float* __restrict__ vr,
                               float* __restrict__ sc, float* __restrict__ sh, int d) {
  int c = blockIdx.x * blockDim.x + threadIdx.x;
  if (c < d) {
    float s = g[c] * rsqrtf(vr[c] + EPSBN);
    sc[c] = s;
    sh[c] = bt[c] - mn[c] * s;
  }
}

// ---------------------------------------------------------------------------
// Elementwise helpers
// ---------------------------------------------------------------------------
__global__ void zero_kernel(float* __restrict__ p, long n) {
  long i = (long)blockIdx.x * blockDim.x + threadIdx.x;
  if (i < n) p[i] = 0.0f;
}

__global__ void count_kernel(const int* __restrict__ tgt, float* __restrict__ cnt,
                             int E) {
  int i = blockIdx.x * blockDim.x + threadIdx.x;
  if (i < E) atomicAdd(&cnt[tgt[i]], 1.0f);
}

__global__ void bn_node_kernel(const float* __restrict__ x, float* __restrict__ y,
                               const float* __restrict__ sc, const float* __restrict__ sh,
                               int n, int d) {
  long i = (long)blockIdx.x * blockDim.x + threadIdx.x;
  long tot = (long)n * d;
  if (i < tot) {
    int c = (int)(i % d);
    y[i] = fmaf(x[i], sc[c], sh[c]);
  }
}

__global__ void div_kernel(const float* __restrict__ acc, const float* __restrict__ cnt,
                           float* __restrict__ y, int n, int d) {
  long i = (long)blockIdx.x * blockDim.x + threadIdx.x;
  long tot = (long)n * d;
  if (i < tot) y[i] = acc[i] / fmaxf(cnt[i / d], 1.0f);
}

// ---------------------------------------------------------------------------
// Fused gather -> 2-layer MLP (WMMA) -> scatter-add message kernel.
// MODE 0: input = cat(xb[tgt], edge_attr[e])          (encoder)
// MODE 1: input = cat(xb[tgt], xb[src] - xb[tgt])     (decoder, dec=True)
// One wave per 16-edge tile, 8 waves (128 edges) per block.
// ---------------------------------------------------------------------------
template <int MODE, int DNODE, int KIN, int DOUT, int RELU>
__global__ void __launch_bounds__(256)
edge_mlp_kernel(const float* __restrict__ xb,
                const float* __restrict__ ea,
                const int* __restrict__ srci,
                const int* __restrict__ tgti,
                const _Float16* __restrict__ P1, const float* __restrict__ b1,
                const _Float16* __restrict__ P2, const float* __restrict__ b2,
                float* __restrict__ acc, int E) {
  constexpr int KCH = (KIN + 31) / 32;            // K chunks of 32
  constexpr int KW  = KCH * 32;                   // padded K width
  __shared__ __align__(16) _Float16 s_in[8][16][KW];
  __shared__ __align__(16) _Float16 s_hid[8][16][32];
  __shared__ int s_tgt[8][16];                    // per-edge target (-1 = invalid)

  const int wave = threadIdx.x >> 5;
  const int lane = threadIdx.x & 31;
  const int base = (blockIdx.x * 8 + wave) * 16;

  __builtin_prefetch(P1, 0, 0);
  __builtin_prefetch(P2, 0, 0);

  // ---- stage input features (f16) into LDS -------------------------------
  for (int e = 0; e < 16; ++e) {
    const int  eid   = base + e;
    const bool valid = eid < E;
    const int  ec    = valid ? eid : (E - 1);
    const int  t     = tgti[ec];
    const int  s     = srci[ec];
    if (lane == 0) s_tgt[wave][e] = valid ? t : -1;
#pragma unroll
    for (int fp = 0; fp < KW; fp += 32) {
      const int f = fp + lane;
      float v = 0.0f;
      if (MODE == 0) {
        if (f < DNODE)      v = xb[(long)t * DNODE + f];
        else if (f < KIN)   v = ea[(long)ec * 3 + (f - DNODE)];
      } else {
        if (f < DNODE)      v = xb[(long)t * DNODE + f];
        else if (f < KIN) {
          const int c = f - DNODE;
          v = xb[(long)s * DNODE + c] - xb[(long)t * DNODE + c];
        }
      }
      s_in[wave][e][f] = (_Float16)v;
    }
  }
  __syncthreads();

  const int mb = (lane >> 4) ? 8 : 0;
  const int n0 = lane & 15;

  // ---- layer 1: KIN -> 32, ReLU ------------------------------------------
  v8f c0 = {}, c1 = {};
#pragma unroll
  for (int kc = 0; kc < KCH; ++kc) {
    v16h a  = load_A(&s_in[wave][0][0], lane, kc * 32, KW);
    v16h bl = load_Bp(P1, kc * 32, 0, lane);
    c0 = wmma16(a, bl, c0);
    v16h bh = load_Bp(P1, kc * 32, 16, lane);
    c1 = wmma16(a, bh, c1);
  }
  {
    const float bia0 = b1[n0], bia1 = b1[n0 + 16];
#pragma unroll
    for (int r = 0; r < 8; ++r) {
      s_hid[wave][mb + r][n0]      = (_Float16)fmaxf(c0[r] + bia0, 0.0f);
      s_hid[wave][mb + r][n0 + 16] = (_Float16)fmaxf(c1[r] + bia1, 0.0f);
    }
  }
  __syncthreads();

  // ---- layer 2: 32 -> DOUT (+optional ReLU), scatter-add -----------------
  v16h a2 = load_A(&s_hid[wave][0][0], lane, 0, 32);
  v8f d0 = {}, d1 = {};
  {
    v16h b20 = load_Bp(P2, 0, 0, lane);
    d0 = wmma16(a2, b20, d0);
  }
  if (DOUT > 16) {
    v16h b21 = load_Bp(P2, 0, 16, lane);
    d1 = wmma16(a2, b21, d1);
  }

#pragma unroll
  for (int r = 0; r < 8; ++r) {
    const int m = mb + r;
    const int t = s_tgt[wave][m];
    if (t >= 0) {
      if (n0 < DOUT) {
        float v = d0[r] + b2[n0];
        if (RELU) v = fmaxf(v, 0.0f);
        atomicAdd(&acc[(long)t * DOUT + n0], v);
      }
      if (DOUT > 16) {
        const int n2 = n0 + 16;
        float v = d1[r] + b2[n2];
        if (RELU) v = fmaxf(v, 0.0f);
        atomicAdd(&acc[(long)t * DOUT + n2], v);
      }
    }
  }
}

// ---------------------------------------------------------------------------
// Edge-attribute predictor: bn1 -> fc1(4->32)+ReLU -> bn2 -> fc2(32->32)
// -> h + bn3(ReLU(h_res)) -> fc4(32->3).  One wave per 16-edge tile.
// All BN layers pre-folded to scale/shift.
// ---------------------------------------------------------------------------
__global__ void __launch_bounds__(256)
edge_pred_kernel(const float* __restrict__ xenc,
                 const int* __restrict__ srci,
                 const int* __restrict__ tgti,
                 const float* __restrict__ sc1, const float* __restrict__ sh1,
                 const _Float16* __restrict__ P1, const float* __restrict__ b1,
                 const float* __restrict__ sc2, const float* __restrict__ sh2,
                 const _Float16* __restrict__ P2, const float* __restrict__ b2,
                 const float* __restrict__ sc3, const float* __restrict__ sh3,
                 const _Float16* __restrict__ P4, const float* __restrict__ b4,
                 float* __restrict__ out, int E) {
  __shared__ __align__(16) _Float16 s_in[8][16][32];
  __shared__ __align__(16) _Float16 s_hid[8][16][32];

  const int wave = threadIdx.x >> 5;
  const int lane = threadIdx.x & 31;
  const int base = (blockIdx.x * 8 + wave) * 16;

  // ---- input: bn1(cat(xenc[src], xenc[tgt])) -----------------------------
  for (int e = 0; e < 16; ++e) {
    const int eid = base + e;
    const int ec  = (eid < E) ? eid : (E - 1);
    const int s = srci[ec], t = tgti[ec];
    const int f = lane;
    float v = 0.0f;
    if (f < 2)      v = xenc[(long)s * 2 + f];
    else if (f < 4) v = xenc[(long)t * 2 + (f - 2)];
    if (f < 4) v = fmaf(v, sc1[f], sh1[f]);
    s_in[wave][e][f] = (_Float16)v;
  }
  __syncthreads();

  const int mb = (lane >> 4) ? 8 : 0;
  const int n0 = lane & 15;
  const int n1 = n0 + 16;

  // ---- fc1 (K=4 padded to 32) + ReLU + bn2 -------------------------------
  {
    v16h a  = load_A(&s_in[wave][0][0], lane, 0, 32);
    v16h bl = load_Bp(P1, 0, 0, lane);
    v16h bh = load_Bp(P1, 0, 16, lane);
    v8f c0 = {}, c1 = {};
    c0 = wmma16(a, bl, c0);
    c1 = wmma16(a, bh, c1);
    const float s0 = sc2[n0], h0s = sh2[n0], s1 = sc2[n1], h1s = sh2[n1];
    const float bb0 = b1[n0], bb1 = b1[n1];
#pragma unroll
    for (int r = 0; r < 8; ++r) {
      float h0 = fmaxf(c0[r] + bb0, 0.0f);
      float h1 = fmaxf(c1[r] + bb1, 0.0f);
      s_hid[wave][mb + r][n0] = (_Float16)fmaf(h0, s0, h0s);
      s_hid[wave][mb + r][n1] = (_Float16)fmaf(h1, s1, h1s);
    }
  }
  __syncthreads();

  // ---- fc2 + residual: h_final = h + bn3(ReLU(h@W2^T + b2)) --------------
  {
    v16h a  = load_A(&s_hid[wave][0][0], lane, 0, 32);
    v16h bl = load_Bp(P2, 0, 0, lane);
    v16h bh = load_Bp(P2, 0, 16, lane);
    v8f c0 = {}, c1 = {};
    c0 = wmma16(a, bl, c0);
    c1 = wmma16(a, bh, c1);
    const float s0 = sc3[n0], h0s = sh3[n0], s1 = sc3[n1], h1s = sh3[n1];
    const float bb0 = b2[n0], bb1 = b2[n1];
#pragma unroll
    for (int r = 0; r < 8; ++r) {
      const int m = mb + r;
      float t0 = fmaxf(c0[r] + bb0, 0.0f);
      float t1 = fmaxf(c1[r] + bb1, 0.0f);
      t0 = fmaf(t0, s0, h0s);
      t1 = fmaf(t1, s1, h1s);
      const float hf0 = (float)s_hid[wave][m][n0] + t0;
      const float hf1 = (float)s_hid[wave][m][n1] + t1;
      s_in[wave][m][n0] = (_Float16)hf0;   // reuse input tile as staging
      s_in[wave][m][n1] = (_Float16)hf1;
    }
  }
  __syncthreads();

  // ---- fc4: 32 -> 3, direct store ----------------------------------------
  {
    v16h a  = load_A(&s_in[wave][0][0], lane, 0, 32);
    v16h b0 = load_Bp(P4, 0, 0, lane);
    v8f d = {};
    d = wmma16(a, b0, d);
#pragma unroll
    for (int r = 0; r < 8; ++r) {
      const int eid = base + mb + r;
      if (eid < E && n0 < 3) out[(long)eid * 3 + n0] = d[r] + b4[n0];
    }
  }
}

// ---------------------------------------------------------------------------
// Host-side orchestration
// ---------------------------------------------------------------------------
enum {
  IN_X = 0, IN_EA = 1,
  E0_G = 2, E0_B, E0_M, E0_V, E0_W1, E0_B1, E0_W2, E0_B2,
  E1_G,     E1_B, E1_M, E1_V, E1_W1, E1_B1, E1_W2, E1_B2,
  D0_G,     D0_B, D0_M, D0_V, D0_W1, D0_B1, D0_W2, D0_B2,
  D1_G,     D1_B, D1_M, D1_V, D1_W1, D1_B1, D1_W2, D1_B2,
  P_G1, P_B1, P_M1, P_V1, P_W1, P_BB1,
  P_G2, P_B2, P_M2, P_V2, P_W2, P_BB2,
  P_G3, P_B3, P_M3, P_V3, P_W4, P_BB4,
  IN_EIDX  // = 52
};

extern "C" void kernel_launch(void* const* d_in, const int* in_sizes, int n_in,
                              void* d_out, int out_size, void* d_ws, size_t ws_size,
                              hipStream_t stream) {
  (void)out_size; (void)ws_size;
  const int N = in_sizes[IN_X] / 3;
  const int E = in_sizes[IN_EA] / 3;

  auto F = [&](int i) { return (const float*)d_in[i]; };
  const float* x  = F(IN_X);
  const float* ea = F(IN_EA);
  const int* eidx = (const int*)d_in[n_in - 1];   // edge_index [2][E]
  const int* srci = eidx;
  const int* tgti = eidx + E;

  // ---- workspace layout ---------------------------------------------------
  float* ws   = (float*)d_ws;
  float* cnt  = ws;                 // [N]
  float* xcur = cnt + N;            // [N*32]
  float* xb   = xcur + (long)N*32;  // [N*32]
  float* acc  = xb   + (long)N*32;  // [N*32]
  float* xenc = acc  + (long)N*32;  // [N*2]
  float* outv = (float*)d_out;      // xr [N*3] then pred_edge_attr [E*3]

  // f16 weight pool (256B aligned)
  uintptr_t pb = (uintptr_t)(xenc + (long)N * 2);
  pb = (pb + 255) & ~(uintptr_t)255;
  _Float16* pool = (_Float16*)pb;
  auto alloc_p = [&](int kpad) { _Float16* p = pool; pool += (long)kpad * 32; return p; };
  _Float16* pe0w1 = alloc_p(32);  _Float16* pe0w2 = alloc_p(32);
  _Float16* pe1w1 = alloc_p(64);  _Float16* pe1w2 = alloc_p(32);
  _Float16* pd0w1 = alloc_p(32);  _Float16* pd0w2 = alloc_p(32);
  _Float16* pd1w1 = alloc_p(64);  _Float16* pd1w2 = alloc_p(32);
  _Float16* ppw1  = alloc_p(32);  _Float16* ppw2  = alloc_p(32);
  _Float16* ppw4  = alloc_p(32);

  // folded-BN scale/shift pool (float, after the f16 pool, 256B aligned)
  uintptr_t bb = (uintptr_t)pool;
  bb = (bb + 255) & ~(uintptr_t)255;
  float* bpool = (float*)bb;
  auto alloc_b = [&](int d) { float* p = bpool; bpool += d; return p; };
  float* e0sc = alloc_b(3);  float* e0sh = alloc_b(3);
  float* e1sc = alloc_b(32); float* e1sh = alloc_b(32);
  float* d0sc = alloc_b(2);  float* d0sh = alloc_b(2);
  float* d1sc = alloc_b(32); float* d1sh = alloc_b(32);
  float* p1sc = alloc_b(4);  float* p1sh = alloc_b(4);
  float* p2sc = alloc_b(32); float* p2sh = alloc_b(32);
  float* p3sc = alloc_b(32); float* p3sh = alloc_b(32);

  const int T = 256;
  auto gN = [&](long n) { return dim3((unsigned)((n + T - 1) / T)); };
  const dim3 gE((E + 127) / 128);   // 8 waves x 16 edges per block

  // ---- pack weights (WMMA B layout) and fold BN params --------------------
  auto pack = [&](int wi, _Float16* p, int kin, int dout, int kpad) {
    pack_w_kernel<<<gN((long)kpad * 32), T, 0, stream>>>(F(wi), p, kin, dout, kpad);
  };
  pack(E0_W1, pe0w1,  6, 32, 32);  pack(E0_W2, pe0w2, 32, 32, 32);
  pack(E1_W1, pe1w1, 35, 32, 64);  pack(E1_W2, pe1w2, 32,  2, 32);
  pack(D0_W1, pd0w1,  4, 32, 32);  pack(D0_W2, pd0w2, 32, 32, 32);
  pack(D1_W1, pd1w1, 64, 32, 64);  pack(D1_W2, pd1w2, 32,  3, 32);
  pack(P_W1,  ppw1,   4, 32, 32);  pack(P_W2,  ppw2,  32, 32, 32);
  pack(P_W4,  ppw4,  32,  3, 32);

  auto foldbn = [&](int gi, float* sc, float* sh, int d) {
    pack_bn_kernel<<<1, 64, 0, stream>>>(F(gi), F(gi + 1), F(gi + 2), F(gi + 3), sc, sh, d);
  };
  foldbn(E0_G, e0sc, e0sh, 3);
  foldbn(E1_G, e1sc, e1sh, 32);
  foldbn(D0_G, d0sc, d0sh, 2);
  foldbn(D1_G, d1sc, d1sh, 32);
  foldbn(P_G1, p1sc, p1sh, 4);
  foldbn(P_G2, p2sc, p2sh, 32);
  foldbn(P_G3, p3sc, p3sh, 32);

  // degree counts
  zero_kernel<<<gN(N), T, 0, stream>>>(cnt, N);
  count_kernel<<<gN(E), T, 0, stream>>>(tgti, cnt, E);

  // ---- encoder block 0: 3 -> 32 ------------------------------------------
  bn_node_kernel<<<gN((long)N*3), T, 0, stream>>>(x, xb, e0sc, e0sh, N, 3);
  zero_kernel<<<gN((long)N*32), T, 0, stream>>>(acc, (long)N*32);
  edge_mlp_kernel<0, 3, 6, 32, 1><<<gE, T, 0, stream>>>(
      xb, ea, srci, tgti, pe0w1, F(E0_B1), pe0w2, F(E0_B2), acc, E);
  div_kernel<<<gN((long)N*32), T, 0, stream>>>(acc, cnt, xcur, N, 32);

  // ---- encoder block 1: 32 -> 2 ------------------------------------------
  bn_node_kernel<<<gN((long)N*32), T, 0, stream>>>(xcur, xb, e1sc, e1sh, N, 32);
  zero_kernel<<<gN((long)N*2), T, 0, stream>>>(acc, (long)N*2);
  edge_mlp_kernel<0, 32, 35, 2, 1><<<gE, T, 0, stream>>>(
      xb, ea, srci, tgti, pe1w1, F(E1_B1), pe1w2, F(E1_B2), acc, E);
  div_kernel<<<gN((long)N*2), T, 0, stream>>>(acc, cnt, xenc, N, 2);

  // ---- decoder block 0: 2 -> 32 (diff messages) --------------------------
  bn_node_kernel<<<gN((long)N*2), T, 0, stream>>>(xenc, xb, d0sc, d0sh, N, 2);
  zero_kernel<<<gN((long)N*32), T, 0, stream>>>(acc, (long)N*32);
  edge_mlp_kernel<1, 2, 4, 32, 1><<<gE, T, 0, stream>>>(
      xb, ea, srci, tgti, pd0w1, F(D0_B1), pd0w2, F(D0_B2), acc, E);
  div_kernel<<<gN((long)N*32), T, 0, stream>>>(acc, cnt, xcur, N, 32);

  // ---- decoder block 1: 32 -> 3 (diff messages, no final ReLU) -----------
  bn_node_kernel<<<gN((long)N*32), T, 0, stream>>>(xcur, xb, d1sc, d1sh, N, 32);
  zero_kernel<<<gN((long)N*3), T, 0, stream>>>(acc, (long)N*3);
  edge_mlp_kernel<1, 32, 64, 3, 0><<<gE, T, 0, stream>>>(
      xb, ea, srci, tgti, pd1w1, F(D1_B1), pd1w2, F(D1_B2), acc, E);
  div_kernel<<<gN((long)N*3), T, 0, stream>>>(acc, cnt, outv, N, 3);  // xr output

  // ---- edge attribute predictor ------------------------------------------
  edge_pred_kernel<<<gE, T, 0, stream>>>(
      xenc, srci, tgti,
      p1sc, p1sh, ppw1, F(P_BB1),
      p2sc, p2sh, ppw2, F(P_BB2),
      p3sc, p3sh, ppw4, F(P_BB4),
      outv + (long)N * 3, E);
}
